// BaselineWormholeRouter_23158463660089
// MI455X (gfx1250) — matrix operand order repaired and checked by reading, not compiled, not run on hardware
//
#include <hip/hip_runtime.h>
#include <hip/hip_bf16.h>

// MI455X / gfx1250 "wormhole router":
//   q = l2n(x@Wq^T+bq), k = l2n(x@Wk^T+bk), v = x@Wv^T+bv
//   scores = q@k^T (diag masked), top8, softmax(top8/0.1), weighted gather of v.
// Both big GEMMs run on the WMMA pipe in bf16 (f32 accum); v stays f32.
// Workspace (assumes ws_size >= ~90MB; everything L2-resident on 192MB L2):
//   [0,16M) x_bf16 | [16,18M) Wq_bf | [18,20M) Wk_bf | [20,22M) Wv_bf
//   [22,38M) q_bf (raw->normalized in place) | [38,54M) k_bf | [54,86M) v_f32

typedef __attribute__((ext_vector_type(16))) __bf16 v16bf;
typedef __attribute__((ext_vector_type(8)))  float  v8f;

union FragU { uint4 u[2]; v16bf v; };

#if defined(__has_builtin)
#  if __has_builtin(__builtin_amdgcn_global_load_async_to_lds_b128)
#    define HAVE_ASYNC_LDS 1
#  endif
#endif
#ifndef HAVE_ASYNC_LDS
#  define HAVE_ASYNC_LDS 0
#endif

#if HAVE_ASYNC_LDS
// Signature revealed by the round-2 diagnostic: param 0 is a (non-const)
// int-vector(4) pointer in the global address space; param 1 is the LDS side.
typedef __attribute__((__vector_size__(4 * sizeof(int)))) int v4i_vec;
typedef v4i_vec __attribute__((address_space(1)))* async_gptr;
typedef v4i_vec __attribute__((address_space(3)))* async_lptr;
#endif

static __device__ __forceinline__ unsigned short f2bf(float f) {
  unsigned u = __builtin_bit_cast(unsigned, f);
  unsigned r = u + 0x7FFFu + ((u >> 16) & 1u);   // round-to-nearest-even
  return (unsigned short)(r >> 16);
}
static __device__ __forceinline__ float bf2f(unsigned short h) {
  unsigned u = ((unsigned)h) << 16;
  return __builtin_bit_cast(float, u);
}
static __device__ __forceinline__ uint2 pack4(float4 f) {
  unsigned lo = (unsigned)f2bf(f.x) | ((unsigned)f2bf(f.y) << 16);
  unsigned hi = (unsigned)f2bf(f.z) | ((unsigned)f2bf(f.w) << 16);
  return make_uint2(lo, hi);
}

// ---- x (4,2049,1024) f32 -> x_bf (8192,1024) bf16, dropping token 0 ----
__global__ __launch_bounds__(256) void cvt_x_bf16(const float* __restrict__ x,
                                                  unsigned short* __restrict__ xb) {
  size_t e = (size_t)blockIdx.x * 256 + threadIdx.x;  // float4 units
  size_t o = e << 2;
  int row = (int)(o >> 10);
  int d   = (int)(o & 1023);
  int b   = row >> 11;
  size_t i = (((size_t)(row + b + 1)) << 10) + d;     // b*2049 + (p+1)
  float4 f = *(const float4*)(x + i);
  *(uint2*)(xb + o) = pack4(f);
}

__global__ __launch_bounds__(256) void cvt_w_bf16(const float* __restrict__ w,
                                                  unsigned short* __restrict__ wb) {
  size_t e = (size_t)blockIdx.x * 256 + threadIdx.x;
  size_t o = e << 2;
  float4 f = *(const float4*)(w + o);
  *(uint2*)(wb + o) = pack4(f);
}

// ---- out[m][n] = sum_k xb[m][k]*wb[n][k] + bias[n] ----
// Wave computes a 16x64 tile: one A fragment feeds 4 WMMAs (4 accumulators).
__global__ __launch_bounds__(256) void gemm_qkv(const unsigned short* __restrict__ xb,
                                                const unsigned short* __restrict__ wb,
                                                const float* __restrict__ bias,
                                                unsigned short* __restrict__ out_bf,
                                                float* __restrict__ out_f32,
                                                int mode) {
  int tid  = threadIdx.x;
  int lane = tid & 31, wid = tid >> 5;
  int wf = blockIdx.x * 8 + wid;          // 8192 waves total
  int tm  = (wf >> 4) << 4;               // 512 M-tiles of 16 rows
  int tn0 = (wf & 15) << 6;               // 16 N-tiles of 64 cols
  int la = lane & 15, hb = lane >> 4;

  const unsigned short* arow = xb + (((size_t)(tm  + la)) << 10);
  const unsigned short* brow = wb + (((size_t)(tn0 + la)) << 10);

  v8f acc[4] = {};
  #pragma unroll 2
  for (int kk = 0; kk < 1024; kk += 32) {
    FragU a;
    const unsigned short* ap = arow + kk + hb * 8;    // A: K..K+7, K+16..K+23 per lane
    a.u[0] = *(const uint4*)(ap);
    a.u[1] = *(const uint4*)(ap + 16);
    __builtin_prefetch(ap + 32, 0, 1);
    #pragma unroll
    for (int t = 0; t < 4; ++t) {
      FragU bfr;
      const unsigned short* bp = brow + (((size_t)t) << 14) + kk + hb * 8;
      bfr.u[0] = *(const uint4*)(bp);
      bfr.u[1] = *(const uint4*)(bp + 16);
      acc[t] = __builtin_amdgcn_wmma_f32_16x16x32_bf16(false, a.v, false, bfr.v,
                                                       (short)0, acc[t], false, false);
    }
  }
  if (mode) {
    #pragma unroll
    for (int t = 0; t < 4; ++t) {
      float bv = bias[tn0 + t * 16 + la];
      #pragma unroll
      for (int i = 0; i < 8; ++i) {
        size_t idx = (((size_t)(tm + i + hb * 8)) << 10) + tn0 + t * 16 + la;
        out_bf[idx] = f2bf(acc[t][i] + bv);
      }
    }
  } else {
    #pragma unroll
    for (int t = 0; t < 4; ++t) {
      float bv = bias[tn0 + t * 16 + la];
      #pragma unroll
      for (int i = 0; i < 8; ++i) {
        size_t idx = (((size_t)(tm + i + hb * 8)) << 10) + tn0 + t * 16 + la;
        out_f32[idx] = acc[t][i] + bv;
      }
    }
  }
}

// ---- row-wise L2 normalize, in place, bf16 ----
__global__ __launch_bounds__(256) void l2norm_bf16(unsigned short* __restrict__ q) {
  __shared__ float red[256];
  int tid = threadIdx.x;
  unsigned short* rp = q + (((size_t)blockIdx.x) << 10);
  float s = 0.f;
  #pragma unroll
  for (int t = 0; t < 4; ++t) { float f = bf2f(rp[tid + (t << 8)]); s += f * f; }
  red[tid] = s; __syncthreads();
  for (int off = 128; off > 0; off >>= 1) {
    if (tid < off) red[tid] += red[tid + off];
    __syncthreads();
  }
  float inv = 1.0f / fmaxf(sqrtf(red[0]), 1e-12f);
  #pragma unroll
  for (int t = 0; t < 4; ++t) {
    int d = tid + (t << 8);
    rp[d] = f2bf(bf2f(rp[d]) * inv);
  }
}

// ---- scores + top8 + softmax + feature gather ----
// Block = 128 query rows x all 2048 keys. Keys staged per 32-key chunk into LDS
// (async-to-LDS pipe, ASYNCcnt) and shared by all 8 waves; each wave keeps
// 2 accumulators (16x32 scores per chunk) and a register-resident sorted top-8.
#define KPAD 1032   // 1024 + 8 bf16 pad: row stride 516 dwords == 4 mod 64 banks

__global__ __launch_bounds__(256) void router_topk(const unsigned short* __restrict__ qn,
                                                   const unsigned short* __restrict__ kn,
                                                   const float* __restrict__ v,
                                                   float* __restrict__ routes,
                                                   float* __restrict__ weights,
                                                   float* __restrict__ feat) {
  __shared__ unsigned short s_k[32][KPAD];   // 66KB staged key chunk
  __shared__ float s_tile[8][16][33];        // padded score tiles
  __shared__ float s_w[128][8];
  __shared__ int   s_i[128][8];

  int tid = threadIdx.x, lane = tid & 31, wid = tid >> 5;
  int b = blockIdx.y;
  int p0blk = blockIdx.x * 128;
  int p0 = p0blk + wid * 16;
  int la = lane & 15, hb = lane >> 4;
  size_t rowbase = (size_t)b * 2048;

  const unsigned short* arow = qn + ((rowbase + p0 + la) << 10);

  float tv[8]; int ti[8];
  #pragma unroll
  for (int j = 0; j < 8; ++j) { tv[j] = -3.0e38f; ti[j] = 0; }
  int myp = p0 + la;

  for (int c = 0; c < 2048; c += 32) {
    __syncthreads();   // previous chunk fully consumed before restaging
    {  // cooperative stage of 32 key rows (64KB) into padded LDS
      const unsigned short* gsrc = kn + ((rowbase + c) << 10);
      #pragma unroll
      for (int i = 0; i < 16; ++i) {
        int flat = tid + i * 256;            // 4096 uint4s
        int key = flat >> 7, off = (flat & 127) << 3;   // 128 uint4 per key row
#if HAVE_ASYNC_LDS
        __builtin_amdgcn_global_load_async_to_lds_b128(
            (async_gptr)(gsrc + (((size_t)key) << 10) + off),
            (async_lptr)(&s_k[key][off]), 0, 0);
#else
        uint4 d = *(const uint4*)(gsrc + (((size_t)key) << 10) + off);
        *(uint4*)(&s_k[key][off]) = d;
#endif
      }
#if HAVE_ASYNC_LDS
      asm volatile("s_wait_asynccnt 0" ::: "memory");
#endif
      __syncthreads();
    }

    v8f acc0 = {}, acc1 = {};
    #pragma unroll 2
    for (int kk = 0; kk < 1024; kk += 32) {
      FragU a, b0, b1;
      const unsigned short* ap = arow + kk + hb * 8;
      a.u[0] = *(const uint4*)(ap);
      a.u[1] = *(const uint4*)(ap + 16);
      const unsigned short* p0b = &s_k[la][kk + hb * 8];
      b0.u[0] = *(const uint4*)(p0b);
      b0.u[1] = *(const uint4*)(p0b + 16);
      const unsigned short* p1b = &s_k[16 + la][kk + hb * 8];
      b1.u[0] = *(const uint4*)(p1b);
      b1.u[1] = *(const uint4*)(p1b + 16);
      acc0 = __builtin_amdgcn_wmma_f32_16x16x32_bf16(false, a.v, false, b0.v,
                                                     (short)0, acc0, false, false);
      acc1 = __builtin_amdgcn_wmma_f32_16x16x32_bf16(false, a.v, false, b1.v,
                                                     (short)0, acc1, false, false);
    }
    // spill 16x32 score tile so each row lands in one lane for top-k
    #pragma unroll
    for (int i = 0; i < 8; ++i) {
      s_tile[wid][i + hb * 8][la]      = acc0[i];
      s_tile[wid][i + hb * 8][16 + la] = acc1[i];
    }
    asm volatile("s_wait_dscnt 0" ::: "memory");   // intra-wave LDS RAW

    if (lane < 16) {
      #pragma unroll
      for (int n = 0; n < 32; ++n) {
        int key = c + n;
        float val = s_tile[wid][lane][n];
        if (key == myp) continue;                   // masked diagonal
        if (val > tv[7]) {                          // sorted-descending insertion
          tv[7] = val; ti[7] = key;
          #pragma unroll
          for (int j = 7; j > 0; --j) {
            if (tv[j] > tv[j - 1]) {
              float tf = tv[j]; tv[j] = tv[j - 1]; tv[j - 1] = tf;
              int tt = ti[j]; ti[j] = ti[j - 1]; ti[j - 1] = tt;
            }
          }
        }
      }
    }
  }

  // softmax over top8 / TEMPERATURE(0.1)
  if (lane < 16) {
    float mx = tv[0];
    float w[8]; float s = 0.f;
    #pragma unroll
    for (int j = 0; j < 8; ++j) { w[j] = __expf((tv[j] - mx) * 10.0f); s += w[j]; }
    float inv = 1.0f / s;
    int r = wid * 16 + la;
    size_t obase = (rowbase + p0blk + r) << 3;
    #pragma unroll
    for (int j = 0; j < 8; ++j) {
      float ww = w[j] * inv;
      routes[obase + j]  = (float)ti[j];
      weights[obase + j] = ww;
      s_w[r][j] = ww;
      s_i[r][j] = ti[j];
    }
  }
  __syncthreads();

  // features: block-cooperative weighted gather of v rows
  const float* vb = v + (rowbase << 10);
  for (int r = 0; r < 128; ++r) {
    float w[8]; int id[8];
    #pragma unroll
    for (int j = 0; j < 8; ++j) { w[j] = s_w[r][j]; id[j] = s_i[r][j]; }
    size_t fb = (rowbase + p0blk + r) << 10;
    #pragma unroll
    for (int t = 0; t < 4; ++t) {
      int d = tid + (t << 8);
      float acc = 0.f;
      #pragma unroll
      for (int j = 0; j < 8; ++j) acc += w[j] * vb[(((size_t)id[j]) << 10) + d];
      feat[fb + d] = acc;
    }
  }
}

extern "C" void kernel_launch(void* const* d_in, const int* in_sizes, int n_in,
                              void* d_out, int out_size, void* d_ws, size_t ws_size,
                              hipStream_t stream) {
  const float* x  = (const float*)d_in[0];
  const float* Wq = (const float*)d_in[1];
  const float* bq = (const float*)d_in[2];
  const float* Wk = (const float*)d_in[3];
  const float* bk = (const float*)d_in[4];
  const float* Wv = (const float*)d_in[5];
  const float* bv = (const float*)d_in[6];

  char* ws = (char*)d_ws;
  unsigned short* xb  = (unsigned short*)(ws);
  unsigned short* wqb = (unsigned short*)(ws + (16ull << 20));
  unsigned short* wkb = (unsigned short*)(ws + (18ull << 20));
  unsigned short* wvb = (unsigned short*)(ws + (20ull << 20));
  unsigned short* qb  = (unsigned short*)(ws + (22ull << 20));
  unsigned short* kb  = (unsigned short*)(ws + (38ull << 20));
  float*          vf  = (float*)        (ws + (54ull << 20));

  float* routesO  = (float*)d_out;
  float* weightsO = routesO + 4 * 2048 * 8;
  float* featO    = weightsO + 4 * 2048 * 8;

  cvt_x_bf16<<<8192, 256, 0, stream>>>(x, xb);
  cvt_w_bf16<<<1024, 256, 0, stream>>>(Wq, wqb);
  cvt_w_bf16<<<1024, 256, 0, stream>>>(Wk, wkb);
  cvt_w_bf16<<<1024, 256, 0, stream>>>(Wv, wvb);

  gemm_qkv<<<1024, 256, 0, stream>>>(xb, wqb, bq, qb, nullptr, 1);
  gemm_qkv<<<1024, 256, 0, stream>>>(xb, wkb, bk, kb, nullptr, 1);
  gemm_qkv<<<1024, 256, 0, stream>>>(xb, wvb, bv, nullptr, vf, 0);

  l2norm_bf16<<<8192, 256, 0, stream>>>(qb);
  l2norm_bf16<<<8192, 256, 0, stream>>>(kb);

  router_topk<<<dim3(16, 4), 256, 0, stream>>>(qb, kb, vf, routesO, weightsO, featO);
}
